// GraphWaterDualEncoder_76879914598735
// MI455X (gfx1250) — matrix operand order repaired and checked by reading, not compiled, not run on hardware
//
#include <hip/hip_runtime.h>
#include <hip/hip_bf16.h>

typedef __attribute__((ext_vector_type(16))) _Float16 v16h;
typedef __attribute__((ext_vector_type(8)))  float    v8f;

#define TPB 256

static inline int cdiv(long long a, long long b) { return (int)((a + b - 1) / b); }

__device__ __forceinline__ float lrelu(float x) { return x > 0.f ? x : 0.2f * x; }

// float atomic max via int/uint ordering trick (works for mixed signs)
__device__ __forceinline__ void atomicMaxF(float* addr, float val) {
    if (val >= 0.f) atomicMax((int*)addr, __float_as_int(val));
    else            atomicMin((unsigned int*)addr, (unsigned int)__float_as_int(val));
}

// ---------------------------------------------------------------------------
// generic fill
__global__ void k_fill(float* __restrict__ p, float v, long long n) {
    long long i = (long long)blockIdx.x * blockDim.x + threadIdx.x;
    if (i < n) p[i] = v;
}

// y[N,32] = x[N,Kin] @ W[32,Kin]^T + b   (Kin = 2 or 3, too small for WMMA)
__global__ void k_lin_small(const float* __restrict__ x, const float* __restrict__ W,
                            const float* __restrict__ b, float* __restrict__ y,
                            int Nn, int Kin) {
    long long i = (long long)blockIdx.x * blockDim.x + threadIdx.x;
    if (i >= (long long)Nn * 32) return;
    int n = (int)(i >> 5), o = (int)(i & 31);
    float acc = b[o];
    for (int k = 0; k < Kin; ++k) acc += x[(size_t)n * Kin + k] * W[o * Kin + k];
    y[i] = acc;
}

// ---------------------------------------------------------------------------
// Y[N,CO] = X[N,K] @ W[CO,K]^T (+bias). One wave computes one 16x16 tile via
// V_WMMA_F32_16X16X32_F16, accumulating f32. K multiple of 32, CO multiple of 16.
// A order used here (not ISA table order): lane half h, elem e -> K = e + 16h
// B order used here (same as A): elem e -> K = e + 16h, N = lane&15
// D layout (f32 16x16):    vgpr v -> M = v + 8h, N = lane&15
template <int K, int CO>
__global__ __launch_bounds__(128) void k_gemm_wmma(
    const float* __restrict__ X, const float* __restrict__ Wt,
    const float* __restrict__ bias, float* __restrict__ Y, int Nn) {
    const int nRowTiles = (Nn + 15) >> 4;
    const int colTiles  = CO >> 4;
    int tile = blockIdx.x * 4 + (threadIdx.x >> 5);
    if (tile >= nRowTiles * colTiles) return;       // uniform per wave: EXEC stays all-ones
    int rowTile = tile / colTiles;
    int colTile = tile - rowTile * colTiles;
    const int lane = threadIdx.x & 31;
    const int half = lane >> 4;
    const int r    = lane & 15;
    int row  = rowTile * 16 + r;
    int rowc = row < Nn ? row : Nn - 1;
    v8f c = {};
    for (int kt = 0; kt < K; kt += 32) {
        v16h a, b;
        const float* xr = X + (size_t)rowc * K + kt;
#pragma unroll
        for (int e = 0; e < 16; ++e) {
            int k = e + 16 * half;
            a[e] = (_Float16)xr[k];
        }
        const float* wr = Wt + (size_t)(colTile * 16 + r) * K + kt + 16 * half;
#pragma unroll
        for (int e = 0; e < 16; ++e) b[e] = (_Float16)wr[e];
        c = __builtin_amdgcn_wmma_f32_16x16x32_f16(false, a, false, b, (short)0, c,
                                                   false, false);
    }
    float bv = bias ? bias[colTile * 16 + r] : 0.f;
    float* yb = Y + (size_t)(rowTile * 16) * CO + colTile * 16 + r;
    if (rowTile * 16 + 16 <= Nn) {
        // full tile (always taken when N % 16 == 0): unguarded, no EXEC churn
#pragma unroll
        for (int v = 0; v < 8; ++v)
            yb[(size_t)(v + 8 * half) * CO] = c[v] + bv;
    } else {
#pragma unroll
        for (int v = 0; v < 8; ++v) {
            int orow = rowTile * 16 + v + 8 * half;
            if (orow < Nn) yb[(size_t)(v + 8 * half) * CO] = c[v] + bv;
        }
    }
}

// ---------------------------------------------------------------------------
// degree count (float, for rsqrt) / in-edge count
__global__ void k_edge_count(const int* __restrict__ dsti, long long E, float* __restrict__ deg) {
    long long i = (long long)blockIdx.x * blockDim.x + threadIdx.x;
    if (i < E) atomicAdd(&deg[dsti[i]], 1.f);
}
__global__ void k_rsqrt_inplace(float* __restrict__ p, int Nn) {
    int i = blockIdx.x * blockDim.x + threadIdx.x;
    if (i < Nn) p[i] = rsqrtf(p[i]);
}

// GCN: self-loop init + bias, then per-edge normalized scatter-add
__global__ void k_gcn_init(const float* __restrict__ xp, const float* __restrict__ dinv,
                           const float* __restrict__ b, float* __restrict__ out, int Nn) {
    long long i = (long long)blockIdx.x * blockDim.x + threadIdx.x;
    if (i >= (long long)Nn * 32) return;
    int n = (int)(i >> 5), cch = (int)(i & 31);
    out[i] = dinv[n] * dinv[n] * xp[i] + b[cch];
}
__global__ void k_gcn_edge(const float* __restrict__ xp, const float* __restrict__ dinv,
                           const int* __restrict__ srci, const int* __restrict__ dsti,
                           long long E, float* __restrict__ out) {
    long long i = (long long)blockIdx.x * blockDim.x + threadIdx.x;
    if (i >= E * 32) return;
    long long e = i >> 5; int cch = (int)(i & 31);
    int s = srci[e], d = dsti[e];
    atomicAdd(&out[(size_t)d * 32 + cch], dinv[s] * dinv[d] * xp[(size_t)s * 32 + cch]);
}

// ---------------------------------------------------------------------------
// graph-LayerNorm: global sum/sumsq then normalize with per-channel affine
__global__ void k_ln_reduce(const float* __restrict__ x, long long M, float* __restrict__ sums) {
    __shared__ float s1[TPB], s2[TPB];
    float a = 0.f, b = 0.f;
    for (long long i = (long long)blockIdx.x * TPB + threadIdx.x; i < M;
         i += (long long)gridDim.x * TPB) {
        float v = x[i]; a += v; b += v * v;
    }
    s1[threadIdx.x] = a; s2[threadIdx.x] = b;
    __syncthreads();
    for (int st = TPB / 2; st > 0; st >>= 1) {
        if ((int)threadIdx.x < st) {
            s1[threadIdx.x] += s1[threadIdx.x + st];
            s2[threadIdx.x] += s2[threadIdx.x + st];
        }
        __syncthreads();
    }
    if (threadIdx.x == 0) { atomicAdd(&sums[0], s1[0]); atomicAdd(&sums[1], s2[0]); }
}
__global__ void k_ln_apply(const float* __restrict__ x, float* __restrict__ y,
                           const float* __restrict__ sums, const float* __restrict__ w,
                           const float* __restrict__ b, int C, long long M, int relu) {
    long long i = (long long)blockIdx.x * blockDim.x + threadIdx.x;
    if (i >= M) return;
    float mu  = sums[0] / (float)M;
    float var = sums[1] / (float)M - mu * mu;
    float inv = rsqrtf(var + 1e-5f);
    int cch = (int)(i % C);
    float v = (x[i] - mu) * inv * w[cch] + b[cch];
    if (relu) v = fmaxf(v, 0.f);
    y[i] = v;
}

// ---------------------------------------------------------------------------
// GAT pieces. xp is [N, H*32]; scores asrc/adst are [N,H].
__global__ void k_gat_scores(const float* __restrict__ xp, const float* __restrict__ avs,
                             const float* __restrict__ avd, float* __restrict__ asrc,
                             float* __restrict__ adst, int Nn, int H) {
    long long i = (long long)blockIdx.x * blockDim.x + threadIdx.x;
    if (i >= (long long)Nn * H) return;
    int n = (int)(i / H), h = (int)(i % H);
    const float* xr = xp + (size_t)n * H * 32 + h * 32;
    float a = 0.f, d = 0.f;
    for (int cch = 0; cch < 32; ++cch) {
        a += xr[cch] * avs[h * 32 + cch];
        d += xr[cch] * avd[h * 32 + cch];
    }
    asrc[i] = a; adst[i] = d;
}
__global__ void k_gat_self_m(const float* __restrict__ asrc, const float* __restrict__ adst,
                             float* __restrict__ m, long long NH) {
    long long i = (long long)blockIdx.x * blockDim.x + threadIdx.x;
    if (i < NH) m[i] = lrelu(asrc[i] + adst[i]);
}
__global__ void k_gat_edge_max(const float* __restrict__ asrc, const float* __restrict__ adst,
                               const int* __restrict__ srci, const int* __restrict__ dsti,
                               long long E, int H, float* __restrict__ m) {
    long long i = (long long)blockIdx.x * blockDim.x + threadIdx.x;
    if (i >= E * H) return;
    long long e = i / H; int h = (int)(i % H);
    int s = srci[e], d = dsti[e];
    atomicMaxF(&m[(size_t)d * H + h], lrelu(asrc[(size_t)s * H + h] + adst[(size_t)d * H + h]));
}
__global__ void k_gat_self_z(const float* __restrict__ asrc, const float* __restrict__ adst,
                             const float* __restrict__ m, float* __restrict__ z, long long NH) {
    long long i = (long long)blockIdx.x * blockDim.x + threadIdx.x;
    if (i < NH) z[i] = expf(lrelu(asrc[i] + adst[i]) - m[i]);
}
__global__ void k_gat_edge_z(const float* __restrict__ asrc, const float* __restrict__ adst,
                             const int* __restrict__ srci, const int* __restrict__ dsti,
                             long long E, int H, const float* __restrict__ m,
                             float* __restrict__ z) {
    long long i = (long long)blockIdx.x * blockDim.x + threadIdx.x;
    if (i >= E * H) return;
    long long e = i / H; int h = (int)(i % H);
    int s = srci[e], d = dsti[e];
    atomicAdd(&z[(size_t)d * H + h],
              expf(lrelu(asrc[(size_t)s * H + h] + adst[(size_t)d * H + h]) - m[(size_t)d * H + h]));
}
__global__ void k_gat_coef(const float* __restrict__ asrc, const float* __restrict__ adst,
                           const int* __restrict__ srci, const int* __restrict__ dsti,
                           long long E, int H, const float* __restrict__ m,
                           const float* __restrict__ z, float* __restrict__ coef) {
    long long i = (long long)blockIdx.x * blockDim.x + threadIdx.x;
    if (i >= E * H) return;
    long long e = i / H; int h = (int)(i % H);
    int s = srci[e], d = dsti[e];
    float ev = lrelu(asrc[(size_t)s * H + h] + adst[(size_t)d * H + h]);
    coef[i] = expf(ev - m[(size_t)d * H + h]) / z[(size_t)d * H + h];
}
__global__ void k_gat_out_init(const float* __restrict__ xp, const float* __restrict__ asrc,
                               const float* __restrict__ adst, const float* __restrict__ m,
                               const float* __restrict__ z, const float* __restrict__ bias,
                               float* __restrict__ out, int Nn, int H) {
    long long i = (long long)blockIdx.x * blockDim.x + threadIdx.x;
    long long M = (long long)Nn * H * 32;
    if (i >= M) return;
    int hc = (int)(i % (H * 32));
    int n  = (int)(i / (H * 32));
    int h  = hc >> 5;
    float e0 = lrelu(asrc[(size_t)n * H + h] + adst[(size_t)n * H + h]);
    float cf = expf(e0 - m[(size_t)n * H + h]) / z[(size_t)n * H + h];
    out[i] = cf * xp[i] + bias[hc];
}
__global__ void k_gat_edge_agg(const float* __restrict__ xp, const float* __restrict__ coef,
                               const int* __restrict__ srci, const int* __restrict__ dsti,
                               long long E, int H, float* __restrict__ out) {
    long long i = (long long)blockIdx.x * blockDim.x + threadIdx.x;
    if (i >= E * H * 32) return;
    long long e = i / (H * 32);
    int hc = (int)(i % (H * 32));
    int h  = hc >> 5;
    int s = srci[e], d = dsti[e];
    atomicAdd(&out[(size_t)d * H * 32 + hc], coef[e * H + h] * xp[(size_t)s * H * 32 + hc]);
}

// ---------------------------------------------------------------------------
// mean conv (no self loops) + residual + relu, in-place on xf
__global__ void k_mean_edge(const float* __restrict__ y, const int* __restrict__ srci,
                            const int* __restrict__ dsti, long long E, float* __restrict__ acc) {
    long long i = (long long)blockIdx.x * blockDim.x + threadIdx.x;
    if (i >= E * 32) return;
    long long e = i >> 5; int cch = (int)(i & 31);
    atomicAdd(&acc[(size_t)dsti[e] * 32 + cch], y[(size_t)srci[e] * 32 + cch]);
}
__global__ void k_mean_fin(const float* __restrict__ acc, const float* __restrict__ cnt,
                           float* __restrict__ xf, int Nn) {
    long long i = (long long)blockIdx.x * blockDim.x + threadIdx.x;
    if (i >= (long long)Nn * 32) return;
    int n = (int)(i >> 5);
    float v = acc[i] / fmaxf(cnt[n], 1.f) + xf[i];
    xf[i] = fmaxf(v, 0.f);
}

// decoder: out[n] = dec_b + [h_struct, h_func] . dec_w
__global__ void k_decode(const float* __restrict__ hs, const float* __restrict__ hf,
                         const float* __restrict__ dw, const float* __restrict__ db,
                         float* __restrict__ out, int Nn) {
    int n = blockIdx.x * blockDim.x + threadIdx.x;
    if (n >= Nn) return;
    float a = db[0];
    for (int cch = 0; cch < 32; ++cch)
        a += hs[(size_t)n * 32 + cch] * dw[cch] + hf[(size_t)n * 32 + cch] * dw[32 + cch];
    out[n] = a;
}

// ---------------------------------------------------------------------------
extern "C" void kernel_launch(void* const* d_in, const int* in_sizes, int n_in,
                              void* d_out, int out_size, void* d_ws, size_t ws_size,
                              hipStream_t stream) {
    (void)n_in; (void)out_size; (void)ws_size;
    const float* x_struct = (const float*)d_in[0];
    const float* x_func   = (const float*)d_in[1];
    const int*   ei       = (const int*)d_in[2];
    const int*   eiu      = (const int*)d_in[3];
    const float* lin0_w  = (const float*)d_in[4];
    const float* lin0_b  = (const float*)d_in[5];
    const float* gcn1_w  = (const float*)d_in[6];
    const float* gcn1_b  = (const float*)d_in[7];
    const float* ln_s_w  = (const float*)d_in[8];
    const float* ln_s_b  = (const float*)d_in[9];
    const float* gcn2_w  = (const float*)d_in[10];
    const float* gcn2_b  = (const float*)d_in[11];
    const float* plin0_w = (const float*)d_in[12];
    const float* plin0_b = (const float*)d_in[13];
    const float* g1_w    = (const float*)d_in[14];
    const float* g1_as   = (const float*)d_in[15];
    const float* g1_ad   = (const float*)d_in[16];
    const float* g1_b    = (const float*)d_in[17];
    const float* ln1_w   = (const float*)d_in[18];
    const float* ln1_b   = (const float*)d_in[19];
    const float* g2_w    = (const float*)d_in[20];
    const float* g2_as   = (const float*)d_in[21];
    const float* g2_ad   = (const float*)d_in[22];
    const float* g2_b    = (const float*)d_in[23];
    const float* ln2_w   = (const float*)d_in[24];
    const float* ln2_b   = (const float*)d_in[25];
    const float* plin1_w = (const float*)d_in[26];
    const float* plin1_b = (const float*)d_in[27];
    const float* dec_w   = (const float*)d_in[28];
    const float* dec_b   = (const float*)d_in[29];

    const long long N  = in_sizes[0] / 2;
    const long long E  = in_sizes[2] / 2;
    const long long EU = in_sizes[3] / 2;
    const int* srci = ei;       const int* dsti = ei + E;
    const int* su   = eiu;      const int* du   = eiu + EU;

    float* ws = (float*)d_ws;
    size_t off = 0;
    auto carve = [&](size_t nf) { float* p = ws + off; off += nf; return p; };
    float* scal = carve(16);
    float* dinv = carve(N);
    float* cnt  = carve(N);
    float* asrc = carve(N * 2);
    float* adst = carve(N * 2);
    float* mbuf = carve(N * 2);
    float* zbuf = carve(N * 2);
    float* coef = carve(E * 2);
    float* xp64 = carve(N * 64);
    float* y64  = carve(N * 64);
    float* xs   = carve(N * 32);
    float* h32  = carve(N * 32);
    float* hstr = carve(N * 32);
    float* xf   = carve(N * 32);
    float* xp2  = carve(N * 32);
    float* y2   = carve(N * 32);
    float* acc  = carve(N * 32);

    const int t32 = cdiv(N, 16) * 2;  // tiles for CO=32
    const int t64 = cdiv(N, 16) * 4;  // tiles for CO=64

    // ------- structural branch -------
    k_lin_small<<<cdiv(N * 32, TPB), TPB, 0, stream>>>(x_struct, lin0_w, lin0_b, xs, (int)N, 2);
    k_fill<<<cdiv(N, TPB), TPB, 0, stream>>>(dinv, 1.f, N);            // self-loop degree
    k_edge_count<<<cdiv(EU, TPB), TPB, 0, stream>>>(du, EU, dinv);
    k_rsqrt_inplace<<<cdiv(N, TPB), TPB, 0, stream>>>(dinv, (int)N);
    // GCN1
    k_gemm_wmma<32, 32><<<cdiv(t32, 4), 128, 0, stream>>>(xs, gcn1_w, (const float*)nullptr, xp64, (int)N);
    k_gcn_init<<<cdiv(N * 32, TPB), TPB, 0, stream>>>(xp64, dinv, gcn1_b, h32, (int)N);
    k_gcn_edge<<<cdiv(EU * 32, TPB), TPB, 0, stream>>>(xp64, dinv, su, du, EU, h32);
    // LN + relu
    k_fill<<<1, 32, 0, stream>>>(scal, 0.f, 2);
    k_ln_reduce<<<512, TPB, 0, stream>>>(h32, N * 32, scal);
    k_ln_apply<<<cdiv(N * 32, TPB), TPB, 0, stream>>>(h32, h32, scal, ln_s_w, ln_s_b, 32, N * 32, 1);
    // GCN2
    k_gemm_wmma<32, 32><<<cdiv(t32, 4), 128, 0, stream>>>(h32, gcn2_w, (const float*)nullptr, xp64, (int)N);
    k_gcn_init<<<cdiv(N * 32, TPB), TPB, 0, stream>>>(xp64, dinv, gcn2_b, hstr, (int)N);
    k_gcn_edge<<<cdiv(EU * 32, TPB), TPB, 0, stream>>>(xp64, dinv, su, du, EU, hstr);

    // ------- functional branch -------
    k_lin_small<<<cdiv(N * 32, TPB), TPB, 0, stream>>>(x_func, plin0_w, plin0_b, xf, (int)N, 3);
    k_fill<<<cdiv(N, TPB), TPB, 0, stream>>>(cnt, 0.f, N);
    k_edge_count<<<cdiv(E, TPB), TPB, 0, stream>>>(dsti, E, cnt);

    for (int blk = 0; blk < 5; ++blk) {
        const float* w1  = g1_w + (size_t)blk * 64 * 32;
        const float* as1 = g1_as + (size_t)blk * 64;
        const float* ad1 = g1_ad + (size_t)blk * 64;
        const float* b1  = g1_b + (size_t)blk * 64;
        const float* l1w = ln1_w + (size_t)blk * 64;
        const float* l1b = ln1_b + (size_t)blk * 64;
        const float* w2  = g2_w + (size_t)blk * 32 * 64;
        const float* as2 = g2_as + (size_t)blk * 32;
        const float* ad2 = g2_ad + (size_t)blk * 32;
        const float* b2  = g2_b + (size_t)blk * 32;
        const float* l2w = ln2_w + (size_t)blk * 32;
        const float* l2b = ln2_b + (size_t)blk * 32;

        // GAT1 (H=2): xp = xf @ w1^T  -> [N,64]
        k_gemm_wmma<32, 64><<<cdiv(t64, 4), 128, 0, stream>>>(xf, w1, (const float*)nullptr, xp64, (int)N);
        k_gat_scores<<<cdiv(N * 2, TPB), TPB, 0, stream>>>(xp64, as1, ad1, asrc, adst, (int)N, 2);
        k_gat_self_m<<<cdiv(N * 2, TPB), TPB, 0, stream>>>(asrc, adst, mbuf, N * 2);
        k_gat_edge_max<<<cdiv(E * 2, TPB), TPB, 0, stream>>>(asrc, adst, srci, dsti, E, 2, mbuf);
        k_gat_self_z<<<cdiv(N * 2, TPB), TPB, 0, stream>>>(asrc, adst, mbuf, zbuf, N * 2);
        k_gat_edge_z<<<cdiv(E * 2, TPB), TPB, 0, stream>>>(asrc, adst, srci, dsti, E, 2, mbuf, zbuf);
        k_gat_coef<<<cdiv(E * 2, TPB), TPB, 0, stream>>>(asrc, adst, srci, dsti, E, 2, mbuf, zbuf, coef);
        k_gat_out_init<<<cdiv(N * 64, TPB), TPB, 0, stream>>>(xp64, asrc, adst, mbuf, zbuf, b1, y64, (int)N, 2);
        k_gat_edge_agg<<<cdiv(E * 64, TPB), TPB, 0, stream>>>(xp64, coef, srci, dsti, E, 2, y64);
        // LN1 + relu (C=64)
        k_fill<<<1, 32, 0, stream>>>(scal, 0.f, 2);
        k_ln_reduce<<<512, TPB, 0, stream>>>(y64, N * 64, scal);
        k_ln_apply<<<cdiv(N * 64, TPB), TPB, 0, stream>>>(y64, y64, scal, l1w, l1b, 64, N * 64, 1);
        // GAT2 (H=1, K=64)
        k_gemm_wmma<64, 32><<<cdiv(t32, 4), 128, 0, stream>>>(y64, w2, (const float*)nullptr, xp2, (int)N);
        k_gat_scores<<<cdiv(N, TPB), TPB, 0, stream>>>(xp2, as2, ad2, asrc, adst, (int)N, 1);
        k_gat_self_m<<<cdiv(N, TPB), TPB, 0, stream>>>(asrc, adst, mbuf, N);
        k_gat_edge_max<<<cdiv(E, TPB), TPB, 0, stream>>>(asrc, adst, srci, dsti, E, 1, mbuf);
        k_gat_self_z<<<cdiv(N, TPB), TPB, 0, stream>>>(asrc, adst, mbuf, zbuf, N);
        k_gat_edge_z<<<cdiv(E, TPB), TPB, 0, stream>>>(asrc, adst, srci, dsti, E, 1, mbuf, zbuf);
        k_gat_coef<<<cdiv(E, TPB), TPB, 0, stream>>>(asrc, adst, srci, dsti, E, 1, mbuf, zbuf, coef);
        k_gat_out_init<<<cdiv(N * 32, TPB), TPB, 0, stream>>>(xp2, asrc, adst, mbuf, zbuf, b2, y2, (int)N, 1);
        k_gat_edge_agg<<<cdiv(E * 32, TPB), TPB, 0, stream>>>(xp2, coef, srci, dsti, E, 1, y2);
        // LN2 (C=32, no relu)
        k_fill<<<1, 32, 0, stream>>>(scal, 0.f, 2);
        k_ln_reduce<<<512, TPB, 0, stream>>>(y2, N * 32, scal);
        k_ln_apply<<<cdiv(N * 32, TPB), TPB, 0, stream>>>(y2, y2, scal, l2w, l2b, 32, N * 32, 0);
        // mean conv + residual + relu -> xf
        k_fill<<<cdiv(N * 32, TPB), TPB, 0, stream>>>(acc, 0.f, N * 32);
        k_mean_edge<<<cdiv(E * 32, TPB), TPB, 0, stream>>>(y2, srci, dsti, E, acc);
        k_mean_fin<<<cdiv(N * 32, TPB), TPB, 0, stream>>>(acc, cnt, xf, (int)N);
    }

    // h_func = xf @ plin1^T + b  (reuse xp2)
    k_gemm_wmma<32, 32><<<cdiv(t32, 4), 128, 0, stream>>>(xf, plin1_w, plin1_b, xp2, (int)N);
    // decode
    k_decode<<<cdiv(N, TPB), TPB, 0, stream>>>(hstr, xp2, dec_w, dec_b, (float*)d_out, (int)N);
}
